// AutogradLoss_56934086476280
// MI455X (gfx1250) — matrix operand order repaired and checked by reading, not compiled, not run on hardware
//
#include <hip/hip_runtime.h>
#include <hip/hip_bf16.h>

typedef __attribute__((ext_vector_type(16))) _Float16     v16h;
typedef __attribute__((ext_vector_type(8)))  float        v8f;
typedef __attribute__((ext_vector_type(4)))  float        f32x4;
typedef __attribute__((ext_vector_type(4)))  unsigned int u32x4;
typedef __attribute__((ext_vector_type(8)))  unsigned int u32x8;

// ---------------------------------------------------------------------------
// Kernel 1: loss = (x-y)^2 and d_loss = 2(x-y), fully vectorized (b128),
// nontemporal so the 1.3 GB streaming footprint doesn't thrash L2.
// ---------------------------------------------------------------------------
__global__ void loss_grad_kernel(const f32x4* __restrict__ x,
                                 const f32x4* __restrict__ y,
                                 f32x4* __restrict__ loss,
                                 f32x4* __restrict__ grad,
                                 int n4) {
  int i = blockIdx.x * blockDim.x + threadIdx.x;
  if (i >= n4) return;
  f32x4 a = __builtin_nontemporal_load(&x[i]);
  f32x4 b = __builtin_nontemporal_load(&y[i]);
  f32x4 d = a - b;
  __builtin_nontemporal_store(d * d, &loss[i]);
  __builtin_nontemporal_store(d + d, &grad[i]);
}

// ---------------------------------------------------------------------------
// Kernel 2: sqd_loss = 2*I per sample. One wave per block:
//   WMMA generates the 16x16 (2*I) f32 tile -> LDS stage -> TDM iterated
//   store streams 256 KB of the constant Hessian region per descriptor.
// ---------------------------------------------------------------------------
__global__ void hessian_tdm_kernel(float* __restrict__ hess) {
  extern __shared__ float tile[];          // 256 floats (1 KB), LDS offset 0 region

  const unsigned lane = threadIdx.x & 31u;
  const unsigned m    = lane & 15u;        // row (A) or col (B/D)
  const unsigned hi   = lane >> 4;         // upper half-wave

  // A = 2*I (16x32 f16). Per ISA A-layout: lane covers K = h + (h>=8?8:0) + hi*8.
  // B = I   (32x16 f16). Same K order as A: K = h + (h>=8?8:0) + hi*8.
  v16h a, b;
#pragma unroll
  for (int h = 0; h < 16; ++h) {
    unsigned Ka = (unsigned)h + ((h & 8) ? 8u : 0u) + (hi ? 8u : 0u);
    unsigned Kb = (unsigned)h + ((h & 8) ? 8u : 0u) + (hi ? 8u : 0u);
    a[h] = (Ka == m) ? (_Float16)2.0f : (_Float16)0.0f;
    b[h] = (Kb == m) ? (_Float16)1.0f : (_Float16)0.0f;
  }

  v8f c = {};
  // D = A x B + 0 = 2*I16 (exact: 2.0 and 1.0 are exact in f16, f32 accum)
  c = __builtin_amdgcn_wmma_f32_16x16x32_f16(false, a, false, b,
                                             (short)0, c, false, false);

  // Scatter D into LDS row-major: VGPR v of lane holds (M = v + 8*hi, N = m)
  const unsigned rbase = hi * 8u;
#pragma unroll
  for (int v = 0; v < 8; ++v)
    tile[(rbase + v) * 16u + m] = c[v];

  // Make LDS writes visible before the TDM engine reads them.
  asm volatile("s_wait_dscnt 0" ::: "memory");

  // Low 32 bits of a generic LDS pointer are the LDS byte offset (aperture rule).
  const unsigned lds_off = (unsigned)(unsigned long long)(uintptr_t)&tile[0];
  const unsigned long long gaddr =
      (unsigned long long)(uintptr_t)(hess + (size_t)blockIdx.x * 65536u); // 256 KB chunk

  // ---- Tensor DMA Descriptor (D#) ----
  // Group 0: count=1 | lds_addr | global_addr[56:0] | type=2
  u32x4 g0;
  g0.x = 1u;                                                    // count=1, user mode
  g0.y = lds_off;                                               // lds_addr (bytes)
  g0.z = (unsigned)gaddr;                                       // global_addr[31:0]
  g0.w = ((unsigned)(gaddr >> 32) & 0x01FFFFFFu) | 0x80000000u; // [56:32] | type=2

  // Group 1: data_size=4B (2), iterate_enable=1; tensor_dim0=256, tensor_dim1=1,
  //          tile_dim0=256, tile_dim1/2=0, tensor_dim0_stride=256.
  u32x8 g1 = { 0x000A0000u,   // dw0: data_size=2<<16 | iterate_enable=1<<19
               0x01000000u,   // dw1: tensor_dim0[15:0]=256 in bits 31:16
               0x00010000u,   // dw2: tensor_dim1[15:0]=1 in bits 31:16
               0x01000000u,   // dw3: tile_dim0=256 in bits 31:16
               0x00000000u,   // dw4: tile_dim1=0, tile_dim2=0
               0x00000100u,   // dw5: tensor_dim0_stride[31:0]=256
               0x00000000u,   // dw6
               0x00000000u }; // dw7

  // Group 2 (iterate mode): tensor_dim2=1, lds_addr_increment=0 (re-read tile),
  //          global_addr_increment=256 elems, iterate_count=255 (256 iters = 256 KB).
  u32x4 g2 = { 1u, 0u, 256u, 0x00FF0000u };
  u32x4 g3 = { 0u, 0u, 0u, 0u };

  // Issue the TDM store directly per CDNA5 ISA (portable across toolchains;
  // avoids the builtin arity mismatch). Operands are SGPR groups 0..3.
  asm volatile("tensor_store_from_lds %0, %1, %2, %3"
               :
               : "s"(g0), "s"(g1), "s"(g2), "s"(g3)
               : "memory");

  __builtin_amdgcn_s_wait_tensorcnt(0);
}

// ---------------------------------------------------------------------------
extern "C" void kernel_launch(void* const* d_in, const int* in_sizes, int n_in,
                              void* d_out, int out_size, void* d_ws, size_t ws_size,
                              hipStream_t stream) {
  const float* x = (const float*)d_in[0];   // model_out [B,16] f32
  const float* y = (const float*)d_in[1];   // y_true    [B,16] f32
  float* out = (float*)d_out;

  const long long BD = in_sizes[0];         // B*D = 16,777,216
  float* loss = out;                        // [B,16]
  float* grad = out + BD;                   // [B,16]
  float* hess = out + 2 * BD;               // [B,16,16] = BD*16 floats

  // Elementwise loss + grad (float4 lanes)
  const int n4  = (int)(BD / 4);
  const int tpb = 256;
  const int blk = (n4 + tpb - 1) / tpb;
  loss_grad_kernel<<<blk, tpb, 0, stream>>>((const f32x4*)x, (const f32x4*)y,
                                            (f32x4*)loss, (f32x4*)grad, n4);

  // Hessian: BD*16 floats total, 65536 floats (256 KB) per block
  const long long hessN   = BD * 16;        // 268,435,456 floats
  const int       tblocks = (int)(hessN / 65536); // 4096
  hessian_tdm_kernel<<<tblocks, 32, 1024, stream>>>(hess);
}
